// PillarMambaEncoder_16733192585334
// MI455X (gfx1250) — compile-verified, matrix operand order checked by reading, not run on hardware
//
#include <hip/hip_runtime.h>
#include <hip/hip_bf16.h>
#include <stdint.h>

typedef __attribute__((ext_vector_type(16))) _Float16     v16h;
typedef __attribute__((ext_vector_type(8)))  float        v8f;
typedef __attribute__((ext_vector_type(8)))  unsigned int v8u;
typedef __attribute__((ext_vector_type(4)))  int          v4i;

#define NK     4096
#define NP     16384
#define CBEV   256
#define BH     496
#define BW     432
#define KSAMP  32
#define RAD2   0.64f
#define NEGV   (-1e9f)
#define CIN    320          // 64 (pointnet) + 256 (bev)
#define COUT   128
#define ROWS   8192         // B * NK

#define CHUNK  2048         // points staged per LDS chunk (double buffered)
#define NKPB   64           // keypoints per block
#define MAXC   64           // max candidates kept per keypoint

#if defined(__has_builtin)
#  if __has_builtin(__builtin_amdgcn_global_load_async_to_lds_b128) && \
      __has_builtin(__builtin_amdgcn_s_wait_asynccnt)
#    define USE_ASYNC_LDS 1
#  endif
#endif
#ifndef USE_ASYNC_LDS
#  define USE_ASYNC_LDS 0
#endif

__device__ __forceinline__ int   iclamp(int v, int lo, int hi){ return v < lo ? lo : (v > hi ? hi : v); }
__device__ __forceinline__ float frelu(float v){ return v > 0.f ? v : 0.f; }

#if USE_ASYNC_LDS
#define GLOBAL_AS __attribute__((address_space(1)))
#define LDS_AS    __attribute__((address_space(3)))
// One 16B lane-granular async DMA transfer: global -> LDS, tracked by ASYNCcnt.
// Builtin signature (per hipcc diagnostic) takes v4i* pointers: global src in
// AS1, LDS dst in AS3. Generic LDS addresses carry the LDS byte offset in
// their low 32 bits, so the
// truncating integer cast yields the AS3 pointer value.
__device__ __forceinline__ void async_copy16(const void* gptr, void* lptr) {
  __builtin_amdgcn_global_load_async_to_lds_b128(
      (GLOBAL_AS v4i*)(uintptr_t)gptr,
      (LDS_AS v4i*)(unsigned int)(uintptr_t)lptr, 0, 0);
}
#endif

// Stage one chunk: [xyz interleaved CHUNK*3 floats][feat CHUNK floats] = 32KB
__device__ __forceinline__ void stage_chunk(float* buf, const float* pts,
                                            const float* pf, int pbase, int tid) {
#if USE_ASYNC_LDS
  const char* gp = (const char*)(pts + (size_t)pbase * 3);
  const char* gf = (const char*)(pf + pbase);
  char* lb = (char*)buf;
  for (int j = tid; j < (CHUNK / 4) * 3; j += 256)      // 24KB of xyz
    async_copy16(gp + (size_t)j * 16, lb + (size_t)j * 16);
  for (int j = tid; j < CHUNK / 4; j += 256)            // 8KB of feats
    async_copy16(gf + (size_t)j * 16, lb + (size_t)CHUNK * 12 + (size_t)j * 16);
#else
  for (int j = tid; j < CHUNK * 3; j += 256) {
    buf[j] = pts[(size_t)pbase * 3 + j];
    if (pbase + CHUNK < NP)
      __builtin_prefetch(pts + (size_t)(pbase + CHUNK) * 3 + j, 0, 1);
  }
  for (int j = tid; j < CHUNK; j += 256) buf[CHUNK * 3 + j] = pf[pbase + j];
#endif
}

__device__ __forceinline__ void wait_async_lds() {
#if USE_ASYNC_LDS
  __builtin_amdgcn_s_wait_asynccnt(0);
#endif
}

// ---------------------------------------------------------------------------
// K0: Wf (320x128 f32, row-major K x N) -> WfT (128x320 f16, N-major) so the
// WMMA B-fragment gather reads consecutive K pairs as dwords.
// ---------------------------------------------------------------------------
__global__ __launch_bounds__(256) void k0_cvt_wf(const float* __restrict__ Wf,
                                                 _Float16* __restrict__ wfT) {
  int i = blockIdx.x * 256 + threadIdx.x;           // over COUT*CIN
  if (i >= CIN * COUT) return;
  int n = i / CIN, k = i - n * CIN;
  wfT[(size_t)n * CIN + k] = (_Float16)Wf[(size_t)k * COUT + n];
}

// ---------------------------------------------------------------------------
// K1: ball query + group + PointNet MLP + max-pool -> feat[:, 0:64] (f16)
// One block = 64 keypoints of one batch. Points double-buffered through LDS
// via async-to-LDS DMA; one wave32 per keypoint scan; lane == sample for the
// MLP/max-pool phase.
// ---------------------------------------------------------------------------
__global__ __launch_bounds__(256)
void k1_ball_mlp(const float* __restrict__ keypoints,
                 const float* __restrict__ points,
                 const float* __restrict__ pfeat,
                 const float* __restrict__ W1, const float* __restrict__ b1,
                 const float* __restrict__ W2, const float* __restrict__ b2,
                 _Float16* __restrict__ featbuf) {
  extern __shared__ char smem_raw[];
  float* buf0 = (float*)smem_raw;                // CHUNK*4 (xyz + feat)
  float* buf1 = buf0 + CHUNK * 4;                // CHUNK*4
  float* w1   = buf1 + CHUNK * 4;                // 4*32
  float* bb1  = w1  + 4 * 32;                    // 32
  float* w2   = bb1 + 32;                        // 32*64
  float* bb2  = w2  + 32 * 64;                   // 64
  float* cd2  = bb2 + 64;                        // NKPB*MAXC
  float* kx   = cd2 + NKPB * MAXC;               // NKPB
  float* ky   = kx  + NKPB;
  float* kz   = ky  + NKPB;
  unsigned int* cidx = (unsigned int*)(kz + NKPB);   // NKPB*MAXC
  int*          ccnt = (int*)(cidx + NKPB * MAXC);   // NKPB

  const int tid   = threadIdx.x;
  const int lane  = tid & 31;
  const int wave  = tid >> 5;
  const int batch = blockIdx.x >> 6;             // 64 blocks per batch
  const int kbase = (blockIdx.x & 63) * NKPB;

  const float* pts = points + (size_t)batch * NP * 3;
  const float* pf  = pfeat  + (size_t)batch * NP;

  // kick off DMA of chunk 0 before doing the (VGPR-path) small setup loads
  stage_chunk(buf0, pts, pf, 0, tid);

  for (int i = tid; i < 4 * 32;  i += 256) w1[i] = W1[i];
  for (int i = tid; i < 32 * 64; i += 256) w2[i] = W2[i];
  if (tid < 32) bb1[tid] = b1[tid];
  if (tid < 64) bb2[tid] = b2[tid];
  if (tid < NKPB) {
    ccnt[tid] = 0;
    const float* kp = keypoints + ((size_t)batch * NK + kbase + tid) * 3;
    kx[tid] = kp[0]; ky[tid] = kp[1]; kz[tid] = kp[2];
  }
  wait_async_lds();
  __syncthreads();

  float* bufs[2] = {buf0, buf1};
  for (int ch = 0; ch < NP / CHUNK; ++ch) {
    const int pbase = ch * CHUNK;
    float* cur = bufs[ch & 1];
    if (ch + 1 < NP / CHUNK)                     // DMA next chunk while scanning
      stage_chunk(bufs[(ch + 1) & 1], pts, pf, (ch + 1) * CHUNK, tid);

    const float* sp = cur;                       // xyz interleaved (stride-3: conflict-free)
    for (int kq = 0; kq < 8; ++kq) {
      const int kp = wave * 8 + kq;
      const float cx = kx[kp], cy = ky[kp], cz = kz[kp];
      for (int i = lane; i < CHUNK; i += 32) {
        float dx = sp[3 * i]     - cx;
        float dy = sp[3 * i + 1] - cy;
        float dz = sp[3 * i + 2] - cz;
        float d2 = dx * dx + dy * dy + dz * dz;
        if (d2 < RAD2) {
          int pos = atomicAdd(&ccnt[kp], 1);
          if (pos < MAXC) {
            cd2[kp * MAXC + pos]  = d2;
            cidx[kp * MAXC + pos] = (unsigned int)(pbase + i);
          }
        }
      }
    }
    wait_async_lds();                            // own DMA landed
    __syncthreads();                             // everyone's DMA + scan done
  }

  // selection (rarely needed: expected in-ball count ~1.6) + MLP + max-pool
  for (int kq = 0; kq < 8; ++kq) {
    const int kp  = wave * 8 + kq;
    int cnt = ccnt[kp]; if (cnt > MAXC) cnt = MAXC;
    int mysel = -1;
    if (cnt <= KSAMP) {
      if (lane < cnt) mysel = lane;
    } else {
      for (int s = 0; s < KSAMP; ++s) {          // extract 32 smallest d2
        float v = 1e30f; int slot = lane;
        if (lane < cnt) v = cd2[kp * MAXC + lane];
        if (lane + 32 < cnt) {
          float v2 = cd2[kp * MAXC + lane + 32];
          if (v2 < v) { v = v2; slot = lane + 32; }
        }
        for (int m = 16; m; m >>= 1) {
          float ov = __shfl_xor(v, m, 32);
          int   os = __shfl_xor(slot, m, 32);
          if (ov < v || (ov == v && os < slot)) { v = ov; slot = os; }
        }
        if (lane == s) mysel = slot;
        if (lane == 0) cd2[kp * MAXC + slot] = 1e30f;
        __threadfence_block();
        __builtin_amdgcn_wave_barrier();
      }
    }

    const bool valid = (mysel >= 0);
    float gx = 0.f, gy = 0.f, gz = 0.f, gf = 0.f;
    if (valid) {
      unsigned int pidx = cidx[kp * MAXC + mysel];
      gx = pts[(size_t)pidx * 3 + 0] - kx[kp];
      gy = pts[(size_t)pidx * 3 + 1] - ky[kp];
      gz = pts[(size_t)pidx * 3 + 2] - kz[kp];
      gf = pf[pidx];
    }
    float h1[32];
#pragma unroll
    for (int j = 0; j < 32; ++j) {
      float a = bb1[j] + gx * w1[0 * 32 + j] + gy * w1[1 * 32 + j]
                       + gz * w1[2 * 32 + j] + gf * w1[3 * 32 + j];
      h1[j] = frelu(a);
    }
    const size_t row = (size_t)batch * NK + kbase + kp;
    for (int c = 0; c < 64; ++c) {
      float acc = bb2[c];
#pragma unroll
      for (int j = 0; j < 32; ++j) acc += h1[j] * w2[j * 64 + c];
      // relu -> mask invalid with NEG -> wave max -> clamp covers "no valid" case
      float mv = valid ? frelu(acc) : NEGV;
      for (int m = 16; m; m >>= 1) { float o = __shfl_xor(mv, m, 32); mv = o > mv ? o : mv; }
      if (lane == 0) featbuf[row * CIN + c] = (_Float16)frelu(mv);
    }
  }
}

// ---------------------------------------------------------------------------
// K2: BEV bilinear interp -> feat[:, 64:320] (f16). thread = (keypoint, ch)
// ---------------------------------------------------------------------------
__global__ __launch_bounds__(256)
void k2_bev(const float* __restrict__ keypoints, const float* __restrict__ bev,
            _Float16* __restrict__ featbuf) {
  int idx = blockIdx.x * 256 + threadIdx.x;
  int kpg = idx >> 8;                 // 0..8191
  int c   = idx & 255;
  int batch = kpg >> 12;
  const float* kp = keypoints + (size_t)kpg * 3;
  float x = (kp[0] - 0.0f)   * (1.0f / 0.16f);
  float y = (kp[1] + 39.68f) * (1.0f / 0.16f);
  int xf = (int)floorf(x), yf = (int)floorf(y);
  int x0 = iclamp(xf, 0, BW - 1), x1 = iclamp(xf + 1, 0, BW - 1);
  int y0 = iclamp(yf, 0, BH - 1), y1 = iclamp(yf + 1, 0, BH - 1);
  float x0f = (float)x0, x1f = (float)x1, y0f = (float)y0, y1f = (float)y1;
  float wa = (x1f - x) * (y1f - y), wb = (x1f - x) * (y - y0f);
  float wc = (x - x0f) * (y1f - y), wd = (x - x0f) * (y - y0f);
  const float* plane = bev + ((size_t)batch * CBEV + c) * (size_t)(BH * BW);
  float v = plane[y0 * BW + x0] * wa + plane[y1 * BW + x0] * wb
          + plane[y0 * BW + x1] * wc + plane[y1 * BW + x1] * wd;
  featbuf[(size_t)kpg * CIN + 64 + c] = (_Float16)v;
}

// ---------------------------------------------------------------------------
// K3: z = feat(8192x320 f16) @ Wf(320x128) via v_wmma_f32_16x16x32_f16.
// One wave per 16x16 output tile; 8 waves/block share one A row-block.
// Fragment gather follows the CDNA5 16-bit A layout: lane owns row (lane&15),
// half-wave (lane>=16) owns K+8; v16h element pair v holds K = 2v (+16 for
// the upper 4 dwords) -> consecutive K pairs load as dwords (compiler fuses
// each fragment into two global_load_b128).
// ---------------------------------------------------------------------------
__global__ __launch_bounds__(256)
void k3_gemm(const _Float16* __restrict__ feat, const _Float16* __restrict__ wfT,
             float* __restrict__ z) {
  const int lane = threadIdx.x & 31;
  const int wave = threadIdx.x >> 5;
  const int row0 = blockIdx.x * 16;   // M tile (512 blocks)
  const int col0 = wave * 16;         // N tile (8 waves)
  const unsigned int* fa = (const unsigned int*)feat;   // row-major, CIN/2 dwords per row
  const unsigned int* fb = (const unsigned int*)wfT;    // n-major,   CIN/2 dwords per col
  const int mr = lane & 15;
  const int hi = (lane >> 4) << 3;    // 0 or 8
  union Frag { v16h h; v8u u; };
  v8f acc = {};
  for (int kb = 0; kb < CIN / 32; ++kb) {
    Frag A, B;
#pragma unroll
    for (int v = 0; v < 8; ++v) {
      int k0 = kb * 32 + ((v & 4) << 2) + hi + ((v & 3) << 1);
      A.u[v] = fa[(size_t)(row0 + mr) * (CIN / 2) + (k0 >> 1)];
      B.u[v] = fb[(size_t)(col0 + mr) * (CIN / 2) + (k0 >> 1)];
    }
    acc = __builtin_amdgcn_wmma_f32_16x16x32_f16(false, A.h, false, B.h,
                                                 (short)0, acc, false, false);
  }
#pragma unroll
  for (int r = 0; r < 8; ++r)
    z[(size_t)(row0 + r + hi) * COUT + (col0 + (lane & 15))] = acc[r];
}

// ---------------------------------------------------------------------------
// K4: per-column mean/var over 8192 rows (deterministic, one block per col)
// ---------------------------------------------------------------------------
__global__ __launch_bounds__(256)
void k4_stats(const float* __restrict__ z, float* __restrict__ stats) {
  __shared__ float s1[256], s2[256];
  const int n = blockIdx.x, tid = threadIdx.x;
  float s = 0.f, q = 0.f;
  for (int i = tid; i < ROWS; i += 256) {
    float v = z[(size_t)i * COUT + n]; s += v; q += v * v;
  }
  s1[tid] = s; s2[tid] = q; __syncthreads();
  for (int off = 128; off; off >>= 1) {
    if (tid < off) { s1[tid] += s1[tid + off]; s2[tid] += s2[tid + off]; }
    __syncthreads();
  }
  if (tid == 0) {
    float mu = s1[0] * (1.0f / ROWS);
    stats[n]        = mu;
    stats[COUT + n] = s2[0] * (1.0f / ROWS) - mu * mu;   // ddof=0, matches jnp.var
  }
}

// ---------------------------------------------------------------------------
// K5: BatchNorm (train) + ReLU -> d_out
// ---------------------------------------------------------------------------
__global__ __launch_bounds__(256)
void k5_bn(const float* __restrict__ z, const float* __restrict__ stats,
           const float* __restrict__ gamma, const float* __restrict__ beta,
           float* __restrict__ out) {
  int i = blockIdx.x * 256 + threadIdx.x;
  int col = i & (COUT - 1);
  float v = (z[i] - stats[col]) * rsqrtf(stats[COUT + col] + 1e-5f) * gamma[col] + beta[col];
  out[i] = frelu(v);
}

extern "C" void kernel_launch(void* const* d_in, const int* in_sizes, int n_in,
                              void* d_out, int out_size, void* d_ws, size_t ws_size,
                              hipStream_t stream) {
  const float* keypoints = (const float*)d_in[0];
  const float* points    = (const float*)d_in[1];
  const float* pfeat     = (const float*)d_in[2];
  const float* bev       = (const float*)d_in[3];
  const float* W1        = (const float*)d_in[4];
  const float* b1        = (const float*)d_in[5];
  const float* W2        = (const float*)d_in[6];
  const float* b2        = (const float*)d_in[7];
  const float* Wf        = (const float*)d_in[8];
  const float* gamma     = (const float*)d_in[9];
  const float* beta      = (const float*)d_in[10];

  char* ws = (char*)d_ws;
  _Float16* featbuf = (_Float16*)ws;                                   // ROWS*CIN f16
  _Float16* wfT     = (_Float16*)(ws + (size_t)ROWS * CIN * 2);        // COUT*CIN f16
  float*    z       = (float*)(ws + (size_t)ROWS * CIN * 2 + (size_t)COUT * CIN * 2);
  float*    stats   = (float*)((char*)z + (size_t)ROWS * COUT * 4);    // 2*COUT f32

  // K1 dynamic LDS: 2 chunk buffers (2*CHUNK*4 floats) + weights (2272 floats)
  //                 + cand d2 + kp xyz + cand idx + counters = 108,416 bytes
  const size_t smem1 = (size_t)(2 * CHUNK * 4 + 2272 + NKPB * MAXC + 3 * NKPB) * 4
                     + (size_t)NKPB * MAXC * 4 + NKPB * 4;

  k0_cvt_wf<<<(CIN * COUT + 255) / 256, 256, 0, stream>>>(Wf, wfT);
  k1_ball_mlp<<<2 * (NK / NKPB), 256, smem1, stream>>>(keypoints, points, pfeat,
                                                       W1, b1, W2, b2, featbuf);
  k2_bev<<<(ROWS * CBEV) / 256, 256, 0, stream>>>(keypoints, bev, featbuf);
  k3_gemm<<<ROWS / 16, 256, 0, stream>>>(featbuf, wfT, z);
  k4_stats<<<COUT, 256, 0, stream>>>(z, stats);
  k5_bn<<<(ROWS * COUT) / 256, 256, 0, stream>>>(z, stats, gamma, beta, (float*)d_out);
}
